// LlamaAttentionCore_87428354277394
// MI455X (gfx1250) — compile-verified, matrix operand order checked by reading, not run on hardware
//
#include <hip/hip_runtime.h>

// ---------------------------------------------------------------------------
// LlamaAttentionCore for MI455X (gfx1250, wave32, WMMA + async global->LDS).
//
// d_out (f32): head_outs[32,512,128] | scaled_keys[8,512,128] |
//              scaled_values[8,512,128].
// Workspace (bf16): Qb[32][512][128], Kb[8][2560][128], Vt[8][128][2560].
//
// Attention kernel: block = 8 waves, one query head, q-tiles qg*8..qg*8+7.
// Per 32-column context step the block async-copies K tile (32x128) and V^T
// tile (128x32) into LDS double buffers (global_load_async_to_lds_b128 /
// ASYNCcnt) while all 8 waves run WMMA on the previous buffer. Computes
// S^T = K x Q^T so P^T's D-register layout feeds O^T = V^T x P^T directly.
// ---------------------------------------------------------------------------

#define NKV    8
#define NH     32
#define HD     128
#define NB     512
#define NCACHE 2048
#define NCTX   2560          // cache + new tokens; padded tail skipped
#define NEGBIG (-1.0e30f)

typedef __attribute__((ext_vector_type(16))) __bf16 v16bf;
typedef __attribute__((ext_vector_type(8)))  __bf16 v8bf;
typedef __attribute__((ext_vector_type(4)))  __bf16 v4bf;
typedef __attribute__((ext_vector_type(8)))  float  v8f;

__device__ __forceinline__ v16bf load_ab16(const __bf16* base) {
  v8bf lo = *(const v8bf*)(base);
  v8bf hi = *(const v8bf*)(base + 16);
  return __builtin_shufflevector(lo, hi, 0,1,2,3,4,5,6,7,8,9,10,11,12,13,14,15);
}

// Async copy 16B global -> LDS (tracked by ASYNCcnt). VDST register holds the
// LDS byte address (relative to the wave's LDS base; dynamic smem starts at 0).
__device__ __forceinline__ void async_ld16(unsigned lds_byte_off,
                                           const __bf16* gptr) {
  asm volatile("global_load_async_to_lds_b128 %0, %1, off"
               :: "v"(lds_byte_off), "v"(gptr)
               : "memory");
}

__device__ __forceinline__ void wait_async0() {
  asm volatile("s_wait_asynccnt 0" ::: "memory");
}

// ---------------------------------------------------------------------------
// Prep kernels
// ---------------------------------------------------------------------------

__global__ void prep_q_kernel(const float* __restrict__ q,
                              unsigned short* __restrict__ qb) {
  int i4 = blockIdx.x * blockDim.x + threadIdx.x;     // NH*NB*HD/4 threads
  float4 v = ((const float4*)q)[i4];
  v4bf p = { (__bf16)v.x, (__bf16)v.y, (__bf16)v.z, (__bf16)v.w };
  ((v4bf*)qb)[i4] = p;
}

__global__ void convert_new_kernel(const float* __restrict__ keys,
                                   const float* __restrict__ values,
                                   const float* __restrict__ kq_scale,
                                   float* __restrict__ out_sk,
                                   float* __restrict__ out_sv,
                                   unsigned short* __restrict__ kb) {
  int i4 = blockIdx.x * blockDim.x + threadIdx.x;     // NKV*NB*HD/4 threads
  float s = kq_scale[0];
  float4 k = ((const float4*)keys)[i4];
  float4 sk = make_float4(k.x * s, k.y * s, k.z * s, k.w * s);
  ((float4*)out_sk)[i4] = sk;
  float4 v = ((const float4*)values)[i4];
  float4 sv = make_float4(fmaxf(v.x, -1e4f), fmaxf(v.y, -1e4f),
                          fmaxf(v.z, -1e4f), fmaxf(v.w, -1e4f));
  ((float4*)out_sv)[i4] = sv;

  int e   = i4 * 4;
  int h   = e / (NB * HD);
  int rem = e - h * (NB * HD);
  int b   = rem >> 7;
  int d   = rem & (HD - 1);
  __bf16* p = (__bf16*)kb + ((size_t)h * NCTX + NCACHE + b) * HD + d;
  v4bf pk = { (__bf16)sk.x, (__bf16)sk.y, (__bf16)sk.z, (__bf16)sk.w };
  *(v4bf*)p = pk;
}

// dst[h][c*pitch + row_off + r] = cvt(op(src[h][r*cols + c])); blockDim (32,8)
__global__ void transpose_cvt_kernel(const float* __restrict__ src,
                                     size_t src_hstride, int cols,
                                     unsigned short* __restrict__ dst,
                                     size_t dst_hstride, int pitch,
                                     int row_off, int do_clamp) {
  __shared__ float tile[32][33];
  const int h = blockIdx.z;
  const float* S = src + (size_t)h * src_hstride;
  __bf16* D = (__bf16*)(dst + (size_t)h * dst_hstride);
  int c0 = blockIdx.x * 32, r0 = blockIdx.y * 32;
#pragma unroll
  for (int j = 0; j < 4; ++j) {
    int r = r0 + threadIdx.y + j * 8;
    tile[threadIdx.y + j * 8][threadIdx.x] =
        S[(size_t)r * cols + c0 + threadIdx.x];
  }
  __syncthreads();
#pragma unroll
  for (int j = 0; j < 4; ++j) {
    int c = c0 + threadIdx.y + j * 8;
    float v = tile[threadIdx.x][threadIdx.y + j * 8];
    if (do_clamp) v = fmaxf(v, -1e4f);
    D[(size_t)c * pitch + row_off + r0 + threadIdx.x] = (__bf16)v;
  }
}

// ---------------------------------------------------------------------------
// Flash attention with LDS double-buffered K/V tiles.
// smem layout (bf16 elems): buf b at b*8192: K tile [32][128] then V [128][32]
// ---------------------------------------------------------------------------
__global__ __launch_bounds__(256)
void attn_kernel(const unsigned short* __restrict__ qb_,
                 const unsigned short* __restrict__ kb_,
                 const unsigned short* __restrict__ vt_,
                 float* __restrict__ out) {
  extern __shared__ __bf16 smem[];

  const __bf16* Qb = (const __bf16*)qb_;
  const __bf16* Kb = (const __bf16*)kb_;
  const __bf16* Vt = (const __bf16*)vt_;

  const int tid    = threadIdx.x;
  const int lane   = tid & 31;
  const int wave   = tid >> 5;
  const int hq     = blockIdx.x >> 2;                  // query head 0..31
  const int qgroup = blockIdx.x & 3;
  const int qtile  = (qgroup << 3) | wave;             // 0..31
  const int hkv    = hq >> 2;
  const int r      = lane & 15;
  const int koff   = (lane >> 4) * 8;
  const int qg     = qtile * 16 + r;                   // this lane's query row
  const int thr    = NCACHE + qg;                      // allowed: pos <= thr

  const __bf16* Kh = Kb + (size_t)hkv * NCTX * HD;
  const __bf16* Vh = Vt + (size_t)hkv * HD * NCTX;

  // Preload Q A-chunks (B operand of S^T wmma).
  const __bf16* Qrow = Qb + ((size_t)hq * NB + qtile * 16 + r) * HD;
  v16bf qa[4];
#pragma unroll
  for (int kc = 0; kc < 4; ++kc) qa[kc] = load_ab16(Qrow + kc * 32 + koff);

  // Block-cooperative async fetch of one 32-column step into buffer `buf`:
  // K tile = 8KB contiguous; V tile = 128 rows x 64B.
  auto issue_tile = [&](int buf, int cb) {
    const unsigned kbyte = (unsigned)buf * 16384u;
    const unsigned vbyte = kbyte + 8192u;
    const __bf16* gk = Kh + (size_t)cb * HD;
    async_ld16(kbyte + (unsigned)tid * 16u,         gk + tid * 8);
    async_ld16(kbyte + (unsigned)(tid + 256) * 16u, gk + (tid + 256) * 8);
    const int j0 = tid, j1 = tid + 256;
    async_ld16(vbyte + (unsigned)j0 * 16u,
               Vh + (size_t)(j0 >> 2) * NCTX + cb + (j0 & 3) * 8);
    async_ld16(vbyte + (unsigned)j1 * 16u,
               Vh + (size_t)(j1 >> 2) * NCTX + cb + (j1 & 3) * 8);
  };

  v8f acc[8] = {};                 // O^T accumulator: 8 dim-chunks of 16
  float mrun = NEGBIG, lrun = 0.0f;

  // Uniform iteration count for the whole block (waves with shorter causal
  // ranges just produce fully-masked tiles: alpha=1, p=0 -> no-op).
  const int nit = 64 + 4 * (qgroup + 1);               // ceil((2048+128g+128)/32)

  issue_tile(0, 0);
  wait_async0();
  __syncthreads();

  for (int it = 0; it < nit; ++it) {
    const int cb  = it << 5;
    const int cur = it & 1;
    if (it + 1 < nit) issue_tile(cur ^ 1, cb + 32);

    const __bf16* lk = smem + (size_t)cur * 8192;      // K tile [32][128]
    const __bf16* lv = lk + 4096;                      // V^T tile [128][32]

    // ---- S^T (2 subtiles of 16 context rows) = K x Q^T over k=128 ----
    v8f s0 = {}, s1 = {};
    const __bf16* Krow0 = lk + r * HD + koff;
    const __bf16* Krow1 = Krow0 + 16 * HD;
#pragma unroll
    for (int kc = 0; kc < 4; ++kc) {
      v16bf ka0 = load_ab16(Krow0 + kc * 32);
      v16bf ka1 = load_ab16(Krow1 + kc * 32);
      s0 = __builtin_amdgcn_wmma_f32_16x16x32_bf16(false, ka0, false, qa[kc],
                                                   (short)0, s0, false, false);
      s1 = __builtin_amdgcn_wmma_f32_16x16x32_bf16(false, ka1, false, qa[kc],
                                                   (short)0, s1, false, false);
    }

    // ---- causal mask (new-token region only; wave-uniform test) ----
    if (cb + 32 > NCACHE) {
#pragma unroll
      for (int i = 0; i < 8; ++i) {
        int c0 = cb + koff + i;          // D layout: VGPR i, lane half
        if (c0 > thr)      s0[i] = NEGBIG;
        if (c0 + 16 > thr) s1[i] = NEGBIG;
      }
    }

    // ---- online softmax (row q lives in lanes q and q+16) ----
    float mloc = s0[0];
#pragma unroll
    for (int i = 0; i < 8; ++i) {
      mloc = fmaxf(mloc, s0[i]);
      mloc = fmaxf(mloc, s1[i]);
    }
    mloc = fmaxf(mloc, __shfl_xor(mloc, 16, 32));
    float mnew  = fmaxf(mrun, mloc);
    float alpha = __expf(mrun - mnew);
    float p0[8], p1[8], ssum = 0.0f;
#pragma unroll
    for (int i = 0; i < 8; ++i) {
      p0[i] = __expf(s0[i] - mnew);
      p1[i] = __expf(s1[i] - mnew);
      ssum += p0[i] + p1[i];
    }
    ssum += __shfl_xor(ssum, 16, 32);
    lrun = lrun * alpha + ssum;
    mrun = mnew;

    // P^T B operand straight from D registers.
    v16bf pb;
#pragma unroll
    for (int i = 0; i < 8; ++i) {
      pb[i]     = (__bf16)p0[i];
      pb[i + 8] = (__bf16)p1[i];
    }

    // ---- O^T = alpha*O^T + V^T x P^T over 8 dim-chunks ----
    const __bf16* Vrow = lv + r * 32 + koff;
#pragma unroll
    for (int m = 0; m < 8; ++m) {
      v16bf va = load_ab16(Vrow + m * 16 * 32);
#pragma unroll
      for (int i = 0; i < 8; ++i) acc[m][i] *= alpha;
      acc[m] = __builtin_amdgcn_wmma_f32_16x16x32_bf16(false, va, false, pb,
                                                       (short)0, acc[m],
                                                       false, false);
    }

    // Rotate double buffers: my async issues done + all waves' reads retired.
    wait_async0();
    __syncthreads();
  }

  // ---- finalize: O[q][d] = O^T[d][q] / l ----
  const float inv = 1.0f / lrun;
  float* orow = out + ((size_t)hq * NB + qg) * HD + koff;
#pragma unroll
  for (int m = 0; m < 8; ++m) {
    float4 v0 = make_float4(acc[m][0] * inv, acc[m][1] * inv,
                            acc[m][2] * inv, acc[m][3] * inv);
    float4 v1 = make_float4(acc[m][4] * inv, acc[m][5] * inv,
                            acc[m][6] * inv, acc[m][7] * inv);
    *(float4*)(orow + m * 16)     = v0;
    *(float4*)(orow + m * 16 + 4) = v1;
  }
}

// ---------------------------------------------------------------------------
extern "C" void kernel_launch(void* const* d_in, const int* in_sizes, int n_in,
                              void* d_out, int out_size, void* d_ws,
                              size_t ws_size, hipStream_t stream) {
  const float* queries = (const float*)d_in[0];
  const float* keys    = (const float*)d_in[1];
  const float* ktc     = (const float*)d_in[2];
  const float* values  = (const float*)d_in[3];
  const float* vcache  = (const float*)d_in[4];
  // d_in[5] = attn_bias: unused, mask computed analytically.
  const float* kqs     = (const float*)d_in[6];

  float* out0   = (float*)d_out;                         // [32,512,128]
  float* out_sk = out0 + (size_t)NH * NB * HD;           // [8,512,128]
  float* out_sv = out_sk + (size_t)NKV * NB * HD;        // [8,512,128]

  unsigned short* Qb = (unsigned short*)d_ws;            // NH*NB*HD bf16
  unsigned short* Kb = Qb + (size_t)NH * NB * HD;        // NKV*NCTX*HD bf16
  unsigned short* Vt = Kb + (size_t)NKV * NCTX * HD;     // NKV*HD*NCTX bf16

  prep_q_kernel<<<(NH * NB * HD / 4) / 256, 256, 0, stream>>>(queries, Qb);
  convert_new_kernel<<<(NKV * NB * HD / 4) / 256, 256, 0, stream>>>(
      keys, values, kqs, out_sk, out_sv, Kb);

  dim3 tb(32, 8);
  transpose_cvt_kernel<<<dim3(NCACHE / 32, HD / 32, NKV), tb, 0, stream>>>(
      ktc, (size_t)HD * NCACHE, NCACHE, Kb, (size_t)NCTX * HD, HD, 0, 0);
  transpose_cvt_kernel<<<dim3(HD / 32, NCACHE / 32, NKV), tb, 0, stream>>>(
      vcache, (size_t)NCACHE * HD, HD, Vt, (size_t)HD * NCTX, NCTX, 0, 0);
  transpose_cvt_kernel<<<dim3(HD / 32, NB / 32, NKV), tb, 0, stream>>>(
      values, (size_t)NB * HD, HD, Vt, (size_t)HD * NCTX, NCTX, NCACHE, 1);

  attn_kernel<<<(NH * (NB / 16)) / 8, 256, 32768, stream>>>(Qb, Kb, Vt, out0);
}
